// MoELayer_10582799417355
// MI455X (gfx1250) — matrix operand order repaired and checked by reading, compiled, not run on hardware
//
#include <hip/hip_runtime.h>
#include <hip/hip_bf16.h>

typedef __bf16 bf16;
typedef __bf16 bf16x16 __attribute__((ext_vector_type(16)));
typedef float  f32x8   __attribute__((ext_vector_type(8)));
typedef unsigned int u32x4 __attribute__((ext_vector_type(4)));
typedef int    i32x8   __attribute__((ext_vector_type(8)));
typedef int    i32x4   __attribute__((ext_vector_type(4)));

#define B_   4
#define N_   1024
#define C_   1024
#define H_   16
#define HID_ 4096

union AFrag { bf16x16 v; unsigned int u[8]; bf16 e[16]; };
union Acc   { f32x8 v; float f[8]; };

static __device__ __forceinline__ f32x8 wmma_bf16(bf16x16 a, bf16x16 b, f32x8 c) {
  // D(16x16 f32) = A(16x32 bf16) x B(32x16 bf16) + C
  return __builtin_amdgcn_wmma_f32_16x16x32_bf16(false, a, false, b, (short)0, c, false, false);
}

// ---------------------------------------------------------------------------
// RMSNorm (f32 in) -> bf16 out (compact rows). Optional segment row mapping
// for expert slices: physical row = (r/width)*N + lo + r%width.
// ---------------------------------------------------------------------------
__global__ __launch_bounds__(256) void rmsnorm_bf16_kernel(
    const float* __restrict__ X, int x_mapped,
    const float* __restrict__ Wt,
    const int* __restrict__ lo_p, const int* __restrict__ hi_p,
    bf16* __restrict__ out)
{
  int lo = lo_p ? *lo_p : 0;
  int hi = hi_p ? *hi_p : N_;
  int width = hi - lo;
  int M = B_ * width;
  int r = blockIdx.x;
  if (r >= M) return;                       // block-uniform early exit
  size_t pr = (size_t)r;
  if (x_mapped) { int q = r / width; pr = (size_t)q * N_ + lo + (r - q * width); }
  const float* xp = X + pr * (size_t)C_;
  int t = threadIdx.x;
  float vals[4]; float s = 0.f;
  #pragma unroll
  for (int j = 0; j < 4; ++j) { float v = xp[t + j * 256]; vals[j] = v; s += v * v; }
  #pragma unroll
  for (int m = 16; m >= 1; m >>= 1) s += __shfl_xor(s, m, 32);
  __shared__ float red[8];
  if ((t & 31) == 0) red[t >> 5] = s;
  __syncthreads();
  float tot = 0.f;
  #pragma unroll
  for (int j = 0; j < 8; ++j) tot += red[j];
  float inv = rsqrtf(tot * (1.0f / C_) + 1e-6f);
  bf16* op = out + (size_t)r * C_;
  #pragma unroll
  for (int j = 0; j < 4; ++j) { int c = t + j * 256; op[c] = (bf16)(vals[j] * inv * Wt[c]); }
}

// ---------------------------------------------------------------------------
// Tiled WMMA GEMM:  Y[M,Nout] = A_bf16[M,K] @ W_f32[Nout,K]^T (+bias)(+SiLU)(+resid)
// 128x128 output tile, BK=32, 256 threads = 8 waves (4 M-waves x 2 N-waves).
// Each wave computes a 32x64 register tile = 8 WMMAs per k-step.
// Double-buffered LDS: one barrier per k-step. W converted f32->bf16 on stage.
// Output / residual rows optionally segment-mapped (in-place expert updates).
// ---------------------------------------------------------------------------
__global__ __launch_bounds__(256) void gemm_wmma_kernel(
    const bf16* __restrict__ A, int a_ld,
    const float* __restrict__ W, int Nout, int K,
    const float* __restrict__ bias,
    const int* __restrict__ lo_p, const int* __restrict__ hi_p,
    float* __restrict__ outF, bf16* __restrict__ outBF,
    const float* __restrict__ resid,
    int o_mapped, int o_ld, int act)
{
  __shared__ __align__(16) bf16 As[2][128][32];
  __shared__ __align__(16) bf16 Bs[2][128][32];
  int lo = lo_p ? *lo_p : 0;
  int hi = hi_p ? *hi_p : N_;
  int width = hi - lo;
  int M = B_ * width;
  int mb = blockIdx.y, nb = blockIdx.x;
  if (mb * 128 >= M) return;                // block-uniform early exit

  int t   = threadIdx.x;
  int wv  = t >> 5, l = t & 31;
  int l15 = l & 15, lhi = l >> 4;
  int wm  = (wv & 3) * 32;                  // wave's M offset in block
  int wn  = (wv >> 2) * 64;                 // wave's N offset in block

  Acc acc[2][4];
  #pragma unroll
  for (int mi = 0; mi < 2; ++mi)
    #pragma unroll
    for (int ni = 0; ni < 4; ++ni)
      #pragma unroll
      for (int i = 0; i < 8; ++i) acc[mi][ni].f[i] = 0.f;

  uint4  areg[2];
  float4 wreg[4];

  auto load_tile = [&](int kk) {
    #pragma unroll
    for (int j = 0; j < 2; ++j) {           // A: 128x32 bf16 = 512 uint4 chunks
      int c = t + 256 * j;
      int gr = mb * 128 + (c >> 2);
      areg[j] = make_uint4(0u, 0u, 0u, 0u);
      if (gr < M) areg[j] = *(const uint4*)(A + (size_t)gr * a_ld + kk + ((c & 3) << 3));
    }
    #pragma unroll
    for (int j = 0; j < 4; ++j) {           // W: 128x32 f32 = 1024 float4 chunks
      int c = t + 256 * j;
      int wr = nb * 128 + (c >> 3);
      wreg[j] = *(const float4*)(W + (size_t)wr * K + kk + ((c & 7) << 2));
    }
  };
  auto store_tile = [&](int buf) {
    #pragma unroll
    for (int j = 0; j < 2; ++j) {
      int c = t + 256 * j;
      *(uint4*)&As[buf][c >> 2][(c & 3) << 3] = areg[j];
    }
    #pragma unroll
    for (int j = 0; j < 4; ++j) {
      int c = t + 256 * j;
      int r = c >> 3, k0 = (c & 7) << 2;
      Bs[buf][r][k0 + 0] = (bf16)wreg[j].x;
      Bs[buf][r][k0 + 1] = (bf16)wreg[j].y;
      Bs[buf][r][k0 + 2] = (bf16)wreg[j].z;
      Bs[buf][r][k0 + 3] = (bf16)wreg[j].w;
    }
  };

  load_tile(0);
  store_tile(0);
  __syncthreads();

  int nsteps = K >> 5;
  for (int s = 0; s < nsteps; ++s) {
    int cur = s & 1;
    bool has_next = (s + 1) < nsteps;
    if (has_next) load_tile((s + 1) << 5);

    AFrag a0, a1, bfr;
    #pragma unroll
    for (int i = 0; i < 8; ++i) {
      int kx = (lhi << 4) + (i << 1);
      a0.u[i] = *(const unsigned int*)&As[cur][wm + l15][kx];
      a1.u[i] = *(const unsigned int*)&As[cur][wm + 16 + l15][kx];
    }
    #pragma unroll
    for (int ni = 0; ni < 4; ++ni) {
      #pragma unroll
      for (int j = 0; j < 8; ++j)
        bfr.u[j] = *(const unsigned int*)&Bs[cur][wn + ni * 16 + l15][(lhi << 4) + (j << 1)];
      acc[0][ni].v = wmma_bf16(a0.v, bfr.v, acc[0][ni].v);
      acc[1][ni].v = wmma_bf16(a1.v, bfr.v, acc[1][ni].v);
    }
    if (has_next) store_tile((s + 1) & 1);
    __syncthreads();
  }

  #pragma unroll
  for (int mi = 0; mi < 2; ++mi) {
    #pragma unroll
    for (int i = 0; i < 8; ++i) {
      int r = mb * 128 + wm + mi * 16 + i + (lhi << 3);
      if (r >= M) continue;
      size_t pr = (size_t)r;
      if (o_mapped) { int q = r / width; pr = (size_t)q * N_ + lo + (r - q * width); }
      #pragma unroll
      for (int ni = 0; ni < 4; ++ni) {
        int col = nb * 128 + wn + ni * 16 + l15;
        float v = acc[mi][ni].f[i];
        if (bias)  v += bias[col];
        if (act)   v = v / (1.0f + __expf(-v));     // SiLU
        if (resid) v += resid[pr * o_ld + col];
        if (outF)  outF[pr * o_ld + col] = v;
        if (outBF) outBF[(size_t)r * o_ld + col] = (bf16)v;
      }
    }
  }
}

// ---------------------------------------------------------------------------
// Flash-style attention. Block = 8 waves = 128 query rows of one (b,h); each
// wave owns 16 rows. K/V 32-key tiles are staged once per block in LDS with
// coalesced b128 loads and shared by all 8 waves. Online softmax; S = Q K^T
// and O += P V via WMMA. qkv layout: [B*N, 3*C] bf16, q|k|v at 0|C|2C.
// ---------------------------------------------------------------------------
__global__ __launch_bounds__(256) void attn_kernel(
    const bf16* __restrict__ qkvb,
    const unsigned char* __restrict__ mask,
    bf16* __restrict__ out)
{
  int b = blockIdx.z, h = blockIdx.y;
  int t = threadIdx.x;
  int wv = t >> 5, l = t & 31;
  int l15 = l & 15, lhi = l >> 4;
  int mbase = blockIdx.x * 128 + wv * 16;
  __shared__ __align__(16) bf16 Ks[32][64];
  __shared__ __align__(16) bf16 Vs[32][64];
  __shared__ __align__(16) bf16 pbuf[8][16][32];

  const size_t ld = 3 * C_;
  const bf16* qp = qkvb + ((size_t)b * N_) * ld + h * 64;
  const bf16* kp = qp + C_;
  const bf16* vp = qp + 2 * C_;

  AFrag qf[2];                                   // Q rows, hd=64 -> two K=32 steps
  {
    int qrow = mbase + l15;
    #pragma unroll
    for (int d = 0; d < 2; ++d)
      #pragma unroll
      for (int i = 0; i < 8; ++i) {
        int kx = d * 32 + (lhi << 4) + (i << 1);
        qf[d].u[i] = *(const unsigned int*)(qp + (size_t)qrow * ld + kx);
      }
  }

  float mrow[8], lrow[8];
  Acc o[4];
  #pragma unroll
  for (int i = 0; i < 8; ++i) { mrow[i] = -1e30f; lrow[i] = 0.f; }
  #pragma unroll
  for (int c = 0; c < 4; ++c)
    #pragma unroll
    for (int i = 0; i < 8; ++i) o[c].f[i] = 0.f;

  int srow = t >> 3;                             // staging: 32 rows x 64 d
  int sd0  = (t & 7) << 3;

  for (int kt = 0; kt < N_ / 32; ++kt) {
    int keybase = kt * 32;
    // Stage K and V tiles (32x64 bf16 each) cooperatively, coalesced b128.
    *(uint4*)&Ks[srow][sd0] = *(const uint4*)(kp + (size_t)(keybase + srow) * ld + sd0);
    *(uint4*)&Vs[srow][sd0] = *(const uint4*)(vp + (size_t)(keybase + srow) * ld + sd0);
    __syncthreads();

    Acc s0, s1;
    #pragma unroll
    for (int i = 0; i < 8; ++i) { s0.f[i] = 0.f; s1.f[i] = 0.f; }
    #pragma unroll
    for (int cn = 0; cn < 2; ++cn) {
      int key = cn * 16 + l15;                   // tile-local key = B col
      #pragma unroll
      for (int d = 0; d < 2; ++d) {
        AFrag kf;                                 // B[d][key] = K[key][d]
        #pragma unroll
        for (int j = 0; j < 8; ++j)
          kf.u[j] = *(const unsigned int*)&Ks[key][d * 32 + (lhi << 4) + (j << 1)];
        if (cn == 0) s0.v = wmma_bf16(qf[d].v, kf.v, s0.v);
        else         s1.v = wmma_bf16(qf[d].v, kf.v, s1.v);
      }
    }
    bool mk0 = mask[(size_t)b * N_ + keybase + l15] != 0;
    bool mk1 = mask[(size_t)b * N_ + keybase + 16 + l15] != 0;
    #pragma unroll
    for (int i = 0; i < 8; ++i) {
      float a0 = mk0 ? s0.f[i] * 0.125f : -1e30f;   // hd^-0.5 = 0.125
      float a1 = mk1 ? s1.f[i] * 0.125f : -1e30f;
      float rmax = fmaxf(a0, a1);
      #pragma unroll
      for (int m = 8; m >= 1; m >>= 1) rmax = fmaxf(rmax, __shfl_xor(rmax, m, 32));
      float mn   = fmaxf(mrow[i], rmax);
      float corr = __expf(mrow[i] - mn);
      float p0 = __expf(a0 - mn);
      float p1 = __expf(a1 - mn);
      float ps = p0 + p1;
      #pragma unroll
      for (int m = 8; m >= 1; m >>= 1) ps += __shfl_xor(ps, m, 32);
      lrow[i] = lrow[i] * corr + ps;
      mrow[i] = mn;
      #pragma unroll
      for (int c = 0; c < 4; ++c) o[c].f[i] *= corr;
      int prow = i + (lhi << 3);                 // C layout -> LDS row-major P
      pbuf[wv][prow][l15]      = (bf16)p0;
      pbuf[wv][prow][16 + l15] = (bf16)p1;
    }
    __syncthreads();
    AFrag pf;                                    // reload P as A fragment
    #pragma unroll
    for (int i = 0; i < 8; ++i) {
      int kx = (lhi << 4) + (i << 1);
      pf.u[i] = *(const unsigned int*)&pbuf[wv][l15][kx];
    }
    #pragma unroll
    for (int c = 0; c < 4; ++c) {
      AFrag vf;                                  // B[k][d] = V[k][d] from LDS
      int dcol = c * 16 + l15;
      int kb2  = lhi << 4;
      #pragma unroll
      for (int j = 0; j < 16; ++j)
        vf.e[j] = Vs[kb2 + j][dcol];
      o[c].v = wmma_bf16(pf.v, vf.v, o[c].v);
    }
    __syncthreads();                             // protects pbuf + Ks/Vs restage
  }
  #pragma unroll
  for (int i = 0; i < 8; ++i) {
    float inv = 1.0f / lrow[i];
    int row = mbase + i + (lhi << 3);
    #pragma unroll
    for (int c = 0; c < 4; ++c)
      out[((size_t)b * N_ + row) * C_ + h * 64 + c * 16 + l15] = (bf16)(o[c].f[i] * inv);
  }
}

// ---------------------------------------------------------------------------
// L2 warm-up: global_prefetch_b8 over the expert weights (~170MB of f32 fits
// the 192MB global L2, so expert GEMMs later stream from L2, not HBM).
// ---------------------------------------------------------------------------
__global__ __launch_bounds__(256) void l2_prefetch_kernel(
    const float* __restrict__ p, unsigned int nfloats)
{
  size_t idx    = (size_t)blockIdx.x * blockDim.x + threadIdx.x;
  size_t stride = (size_t)gridDim.x * blockDim.x;
  for (size_t f = idx * 64; f < nfloats; f += stride * 64)
    __builtin_prefetch(p + f, 0, 1);
}

// ---------------------------------------------------------------------------
// TDM warm: pull one 64x64 f32 weight tile into LDS via the Tensor Data Mover
// (2D descriptor per CDNA5 ISA), wait on TENSORcnt.
// This toolchain exposes the 6-arg builtin: (g0, g1, g2, g3, g_extra, cpol).
// ---------------------------------------------------------------------------
__global__ __launch_bounds__(32) void tdm_l2_warm_kernel(
    const float* __restrict__ Wm, float* __restrict__ sink)
{
  __shared__ __align__(16) char tile[16384];
  unsigned long long ga = (unsigned long long)Wm;
  u32x4 g0;
  g0[0] = 1u;                                              // count=1
  g0[1] = 0u;                                              // lds_addr
  g0[2] = (unsigned int)(ga & 0xFFFFFFFFull);              // global_addr[31:0]
  g0[3] = (unsigned int)((ga >> 32) & 0x1FFFFFFull) | (2u << 30); // addr hi | type=2
  i32x8 g1;
  g1[0] = (2 << 16);                                       // data_size = 4B
  g1[1] = (int)((1024u & 0xFFFFu) << 16);                  // tensor_dim0 lo16
  g1[2] = (int)((4096u & 0xFFFFu) << 16);                  // dim0 hi | tensor_dim1 lo16
  g1[3] = (int)(64u << 16);                                // dim1 hi | tile_dim0=64
  g1[4] = 64;                                              // tile_dim1=64, tile_dim2=0
  g1[5] = 1024;                                            // tensor_dim0_stride lo32
  g1[6] = 0;
  g1[7] = 0;
  i32x4 z4; z4[0] = 0; z4[1] = 0; z4[2] = 0; z4[3] = 0;
  i32x8 z8;
  #pragma unroll
  for (int i = 0; i < 8; ++i) z8[i] = 0;
  __builtin_amdgcn_tensor_load_to_lds(g0, g1, z4, z4, z8, 0);
  __builtin_amdgcn_s_wait_tensorcnt(0);
  __syncthreads();
  volatile char* vb = tile;                                // keep LDS live
  if (threadIdx.x == 0 && vb[0] == (char)0x5B) sink[0] = 1.0f;
}

// ---------------------------------------------------------------------------
extern "C" void kernel_launch(void* const* d_in, const int* in_sizes, int n_in,
                              void* d_out, int out_size, void* d_ws, size_t ws_size,
                              hipStream_t stream) {
  (void)in_sizes; (void)n_in; (void)out_size; (void)ws_size;
  const float*         x          = (const float*)d_in[0];
  const unsigned char* mask       = (const unsigned char*)d_in[1];
  const float*         norm_att_w = (const float*)d_in[2];
  const float*         qkv_w      = (const float*)d_in[3];
  const float*         proj_w     = (const float*)d_in[4];
  const float*         proj_b     = (const float*)d_in[5];
  // expert param blocks in dict order [sp,tm,vs,cp,hs], 5 tensors each
  const int SP = 6, TM = 11, VS = 16, CP = 21, HS = 26;
  const int* idx_vis      = (const int*)d_in[31];
  const int* idx_spatial  = (const int*)d_in[32];
  const int* idx_temporal = (const int*)d_in[33];
  const int* idx_caption  = (const int*)d_in[34];
  const int* idx_history  = (const int*)d_in[35];
  const int* idx_eos      = (const int*)d_in[36];

  float* out = (float*)d_out;
  char* wsb = (char*)d_ws;
  size_t off = 0;
  auto walloc = [&](size_t bytes) -> char* {
    char* p = wsb + off; off += (bytes + 255) & ~(size_t)255; return p;
  };
  bf16*  xn    = (bf16*)walloc((size_t)4096 * 1024 * 2);  // normed x
  bf16*  qkvb  = (bf16*)walloc((size_t)4096 * 3072 * 2);  // fused qkv
  bf16*  attnb = (bf16*)walloc((size_t)4096 * 1024 * 2);  // attention out
  bf16*  hbuf  = (bf16*)walloc((size_t)4096 * 1024 * 2);  // expert normed in
  bf16*  gbuf  = (bf16*)walloc((size_t)4096 * 4096 * 2);  // expert fc1 out
  float* sink  = (float*)walloc(256);

  // Warm global L2 with expert weights while attention path runs.
  for (int e = 0; e < 5; ++e) {
    int base = 6 + 5 * e;
    l2_prefetch_kernel<<<512, 256, 0, stream>>>((const float*)d_in[base + 1], (unsigned)(HID_ * C_));
    l2_prefetch_kernel<<<512, 256, 0, stream>>>((const float*)d_in[base + 3], (unsigned)(C_ * HID_));
  }
  tdm_l2_warm_kernel<<<1, 32, 0, stream>>>((const float*)d_in[SP + 1], sink);

  // Attention path.
  rmsnorm_bf16_kernel<<<4096, 256, 0, stream>>>(x, 0, norm_att_w, nullptr, nullptr, xn);
  gemm_wmma_kernel<<<dim3(24, 32), 256, 0, stream>>>(xn, 1024, qkv_w, 3072, 1024, nullptr,
                                                     nullptr, nullptr,
                                                     nullptr, qkvb, nullptr, 0, 3072, 0);
  attn_kernel<<<dim3(8, 16, 4), 256, 0, stream>>>(qkvb, mask, attnb);
  gemm_wmma_kernel<<<dim3(8, 32), 256, 0, stream>>>(attnb, 1024, proj_w, 1024, 1024, proj_b,
                                                    nullptr, nullptr,
                                                    out, nullptr, x, 0, 1024, 0);

  // Experts, in the reference's sequential order (slices overlap).
  struct Exp { int wbase; const int* lo; const int* hi; };
  Exp exps[5] = {
    { SP, idx_spatial,  idx_temporal },
    { CP, idx_caption,  idx_history  },
    { TM, idx_temporal, idx_caption  },
    { VS, idx_vis,      idx_caption  },
    { HS, idx_history,  idx_eos      },
  };
  for (int e = 0; e < 5; ++e) {
    const float* nw = (const float*)d_in[exps[e].wbase + 0];
    const float* w1 = (const float*)d_in[exps[e].wbase + 1];
    const float* b1 = (const float*)d_in[exps[e].wbase + 2];
    const float* w2 = (const float*)d_in[exps[e].wbase + 3];
    const float* b2 = (const float*)d_in[exps[e].wbase + 4];
    rmsnorm_bf16_kernel<<<4096, 256, 0, stream>>>(out, 1, nw, exps[e].lo, exps[e].hi, hbuf);
    gemm_wmma_kernel<<<dim3(32, 32), 256, 0, stream>>>(hbuf, 1024, w1, HID_, 1024, b1,
                                                       exps[e].lo, exps[e].hi,
                                                       nullptr, gbuf, nullptr, 0, HID_, 1);
    gemm_wmma_kernel<<<dim3(8, 32), 256, 0, stream>>>(gbuf, HID_, w2, 1024, HID_, b2,
                                                      exps[e].lo, exps[e].hi,
                                                      out, nullptr, out, 1, 1024, 0);
  }
}
